// LSTMNetwork_75522704933308
// MI455X (gfx1250) — matrix operand order repaired and checked by reading, not compiled, not run on hardware
//
#include <hip/hip_runtime.h>

typedef __attribute__((ext_vector_type(16))) _Float16 v16h;
typedef __attribute__((ext_vector_type(8)))  _Float16 v8h;
typedef __attribute__((ext_vector_type(8)))  float    v8f;

#define B_TOT 4096
#define T_LEN 512
#define HD    16

// gfx1250 hardware transcendental V_TANH_F32 (probe-confirmed in round 2 asm).
#if __has_builtin(__builtin_amdgcn_tanhf)
__device__ __forceinline__ float ftanh(float x) { return __builtin_amdgcn_tanhf(x); }
#else
__device__ __forceinline__ float ftanh(float x) {
  return 2.0f * __builtin_amdgcn_rcpf(1.0f + __builtin_amdgcn_exp2f(-2.88539008177792681472f * x)) - 1.0f;
}
#endif

template <int N> struct IC { static constexpr int val = N; };

// One wave (32 lanes) owns a 16-row batch tile; the 3 LSTM layers run in a
// SKEWED pipeline: iteration tt computes layer0(tt), layer1(tt-1), layer2(tt-2),
// which are mutually independent -> 6x ds_load_b128, 12x WMMA and the tanh
// chains of all three layers interleave in one block, hiding LDS/trans latency.
// Per layer-step: gates[16,64] = A[16,32] x B[32,64], A = [layer_input | h],
// via 4x v_wmma_f32_16x16x32_f16 with literal-0 C; biases folded into the
// activation fmas (sigmoid(x+b) = 0.5*tanh(0.5x + 0.5b) + 0.5).
__global__ __launch_bounds__(32) void lstm3_wmma(
    const float* __restrict__ x,
    const float* __restrict__ wih0, const float* __restrict__ whh0,
    const float* __restrict__ bih0, const float* __restrict__ bhh0,
    const float* __restrict__ wih1, const float* __restrict__ whh1,
    const float* __restrict__ bih1, const float* __restrict__ bhh1,
    const float* __restrict__ wih2, const float* __restrict__ whh2,
    const float* __restrict__ bih2, const float* __restrict__ bhh2,
    float* __restrict__ out)
{
  // Per-layer staging tile: 16 rows (M) x 32 cols (K): [0..15]=input, [16..31]=h
  __shared__ __align__(16) _Float16 stage[3][16][32];

  const int lane   = threadIdx.x & 31;
  const int n      = lane & 15;        // N index / A-row M
  const int kh     = lane >> 4;        // lane-half select (K striping)
  const int b_base = blockIdx.x * 16;

  // Zero staging: h(-1)=0, and padding K-columns of layer 0 stay 0 forever.
  for (int i = lane; i < 3 * 16 * 32; i += 32)
    (&stage[0][0][0])[i] = (_Float16)0.0f;

  const float* wih[3] = {wih0, wih1, wih2};
  const float* whh[3] = {whh0, whh1, whh2};
  const float* bih[3] = {bih0, bih1, bih2};
  const float* bhh[3] = {bhh0, bhh1, bhh2};

  // Pre-gather B fragments (32x16 f16 tiles, K x N) per ISA 16-bit B layout:
  // element e of v16h holds row K = kh*8 + e (e<8) or 16 + kh*8 + (e-8) (e>=8), column N = lane%16.
  // Rows 0..15 = W_ih^T (layer0: only row 0 = w_ih0), rows 16..31 = W_hh^T.
  v16h  bfrag[3][4];
  float hb[3][4];   // 0.5*(b_ih+b_hh) for sigmoid gates (tiles 0,1,3)
  float gb[3];      //      b_ih+b_hh  for cell gate     (tile 2)
#pragma unroll
  for (int l = 0; l < 3; ++l) {
#pragma unroll
    for (int g = 0; g < 4; ++g) {
      const int gidx = g * 16 + n;                 // gate row (i,f,g,o blocks)
      const float bsum = bih[l][gidx] + bhh[l][gidx];
      hb[l][g] = 0.5f * bsum;
      if (g == 2) gb[l] = bsum;
      v16h bf;
#pragma unroll
      for (int e = 0; e < 16; ++e) {
        const int K = (e < 8) ? (kh * 8 + e) : (16 + kh * 8 + (e - 8));
        float wv;
        if (K < 16) {
          wv = (l == 0) ? ((K == 0) ? wih[0][gidx] : 0.0f)
                        : wih[l][gidx * 16 + K];
        } else {
          wv = whh[l][gidx * 16 + (K - 16)];
        }
        bf[e] = (_Float16)wv;
      }
      bfrag[l][g] = bf;
    }
  }

  // Cell state in registers, C/D layout: element r -> (M = r + kh*8, N = n)
  float cst[3][8];
#pragma unroll
  for (int l = 0; l < 3; ++l)
#pragma unroll
    for (int r = 0; r < 8; ++r) cst[l][r] = 0.0f;

  // --- per-layer phase helpers (all inlined; l is compile-time) -------------
  auto storex = [&](int t) {                      // inject x_t into layer0 A
    if (lane < 16) {
      const int tc = (t < T_LEN) ? t : (T_LEN - 1);
      stage[0][lane][0] = (_Float16)x[(size_t)(b_base + lane) * T_LEN + tc];
    }
  };

  auto loadA = [&](auto LC) -> v16h {             // 2x ds_load_b128
    constexpr int l = decltype(LC)::val;
    const _Float16* row = &stage[l][n][0];
    v8h lo = *(const v8h*)(row + kh * 8);         // K = kh*8 + 0..7
    v8h hi = *(const v8h*)(row + 16 + kh * 8);    // K = 16 + kh*8 + 0..7
    v16h a;
#pragma unroll
    for (int e = 0; e < 8; ++e) { a[e] = lo[e]; a[8 + e] = hi[e]; }
    return a;
  };

  auto gemm = [&](auto LC, v16h a, v8f* acc) {    // 4x v_wmma, literal-0 C
    constexpr int l = decltype(LC)::val;
#pragma unroll
    for (int g = 0; g < 4; ++g) {
      v8f zc = {};                                // lowers to inline SRC2 = 0
      acc[g] = __builtin_amdgcn_wmma_f32_16x16x32_f16(
          false, a, false, bfrag[l][g], (short)0, zc, false, false);
    }
  };

  auto actstore = [&](auto LC, const v8f* acc, bool doOut) {
    constexpr int l = decltype(LC)::val;
    const int lnext = (l < 2) ? (l + 1) : l;      // avoid OOB index in dead code
#pragma unroll
    for (int r = 0; r < 8; ++r) {                 // tiles: 0=i 1=f 2=g 3=o
      // sigmoid(x+b) = 0.5*tanh(0.5x + hb) + 0.5   (bias free via fma)
      const float ti = ftanh(0.5f * acc[0][r] + hb[l][0]);
      const float tf = ftanh(0.5f * acc[1][r] + hb[l][1]);
      const float to = ftanh(0.5f * acc[3][r] + hb[l][3]);
      const float gv = ftanh(acc[2][r] + gb[l]);
      const float is = 0.5f * ti + 0.5f;
      const float fs = 0.5f * tf + 0.5f;
      const float os = 0.5f * to + 0.5f;
      const float c  = fs * cst[l][r] + is * gv;
      cst[l][r] = c;
      const float h = os * ftanh(c);              // -> v_fma_mixlo_f16 on store
      const int  m  = r + kh * 8;                 // batch row in tile
      stage[l][m][16 + n] = (_Float16)h;          // recurrent h for next step
      if (l < 2) {
        stage[lnext][m][n] = (_Float16)h;         // feed next layer (next iter)
      } else if (doOut) {
        out[(size_t)(b_base + m) * HD + n] = h;   // final h of layer 2
      }
    }
  };

  // --- skewed pipeline ------------------------------------------------------
  storex(0);

  { // tt = 0 : fill (layer 0 only)
    v16h a0 = loadA(IC<0>{});
    v8f acc0[4]; gemm(IC<0>{}, a0, acc0);
    actstore(IC<0>{}, acc0, false);
    storex(1);
  }
  { // tt = 1 : fill (layers 0,1)
    v16h a0 = loadA(IC<0>{});
    v16h a1 = loadA(IC<1>{});
    v8f acc0[4], acc1[4];
    gemm(IC<0>{}, a0, acc0); gemm(IC<1>{}, a1, acc1);
    actstore(IC<0>{}, acc0, false); actstore(IC<1>{}, acc1, false);
    storex(2);
  }

  // steady state: all three layer-steps independent -> full interleave
  for (int tt = 2; tt < T_LEN; ++tt) {
    v16h a0 = loadA(IC<0>{});
    v16h a1 = loadA(IC<1>{});
    v16h a2 = loadA(IC<2>{});
    v8f acc0[4], acc1[4], acc2[4];
    gemm(IC<0>{}, a0, acc0);
    gemm(IC<1>{}, a1, acc1);
    gemm(IC<2>{}, a2, acc2);
    actstore(IC<0>{}, acc0, false);
    actstore(IC<1>{}, acc1, false);
    actstore(IC<2>{}, acc2, false);
    storex(tt + 1);                                // clamped inside
  }

  { // tt = T : drain (layers 1,2)
    v16h a1 = loadA(IC<1>{});
    v16h a2 = loadA(IC<2>{});
    v8f acc1[4], acc2[4];
    gemm(IC<1>{}, a1, acc1); gemm(IC<2>{}, a2, acc2);
    actstore(IC<1>{}, acc1, false); actstore(IC<2>{}, acc2, false);
  }
  { // tt = T+1 : drain (layer 2, t = T-1) -> write output
    v16h a2 = loadA(IC<2>{});
    v8f acc2[4]; gemm(IC<2>{}, a2, acc2);
    actstore(IC<2>{}, acc2, true);
  }
}

extern "C" void kernel_launch(void* const* d_in, const int* in_sizes, int n_in,
                              void* d_out, int out_size, void* d_ws, size_t ws_size,
                              hipStream_t stream) {
  (void)in_sizes; (void)n_in; (void)d_ws; (void)ws_size; (void)out_size;
  const float* x    = (const float*)d_in[0];
  const float* wih0 = (const float*)d_in[1];
  const float* whh0 = (const float*)d_in[2];
  const float* bih0 = (const float*)d_in[3];
  const float* bhh0 = (const float*)d_in[4];
  const float* wih1 = (const float*)d_in[5];
  const float* whh1 = (const float*)d_in[6];
  const float* bih1 = (const float*)d_in[7];
  const float* bhh1 = (const float*)d_in[8];
  const float* wih2 = (const float*)d_in[9];
  const float* whh2 = (const float*)d_in[10];
  const float* bih2 = (const float*)d_in[11];
  const float* bhh2 = (const float*)d_in[12];

  lstm3_wmma<<<B_TOT / 16, 32, 0, stream>>>(
      x, wih0, whh0, bih0, bhh0, wih1, whh1, bih1, bhh1,
      wih2, whh2, bih2, bhh2, (float*)d_out);
}
